// JaT_AudioSR_V3_70815420776980
// MI455X (gfx1250) — compile-verified, matrix operand order checked by reading, not run on hardware
//
#include <hip/hip_runtime.h>
#include <cstdint>

#define DEV __device__ __forceinline__

typedef __bf16 bf;
typedef __attribute__((ext_vector_type(16))) __bf16 bf16x16;
typedef __attribute__((ext_vector_type(8)))  float  f32x8;

// ---------------- model dims ----------------
constexpr int BB   = 2;
constexpr int CIN_ = 1024;
constexpr int T_   = 4096;
constexpr int PP   = 4;
constexpr int NN   = 1024;      // tokens = T/P
constexpr int HID_ = 1024;
constexpr int NQ_  = 16;
constexpr int NKV_ = 4;
constexpr int HD_  = 64;
constexpr int MLP_ = 4096;
constexpr int BOT_ = 512;
constexpr float EPS_ = 1e-6f;

// ---------------- gemm tiling ----------------
constexpr int BLK_M = 256;   // 8 waves x 32 rows
constexpr int BLK_N = 64;    // 4 wmma tiles wide
constexpr int KSTEP = 32;

enum { OUT_BF16 = 1, HAS_BIAS = 2, DO_GELU = 4, OUT_TRANS = 8 };

// ---------------- CDNA5 async global->LDS ----------------
DEV void async_copy_b128(uint64_t gaddr, uint32_t lds_off) {
  asm volatile("global_load_async_to_lds_b128 %0, %1, off"
               :: "v"(lds_off), "v"(gaddr) : "memory");
}
DEV void wait_async() {
  asm volatile("s_wait_asynccnt 0" ::: "memory");
}

DEV float gelu_exact(float v) { return 0.5f * v * (1.f + erff(v * 0.70710678118f)); }
DEV float silu(float v) { return v / (1.f + __expf(-v)); }

// ---------------- WMMA GEMM core ----------------
// Computes C[m0:m0+256, n0:n0+64] = A @ B(^T-stored) where
//   A element (m,k) = A[m*lda + k]        (row-major MxK)
//   B element (k,n) = Bm[n*ldb + k]       (B stored transposed: NxK row-major)
// Weights are pre-transposed at conversion; V is produced transposed; K/Q for
// attention scores are naturally in this form.
DEV void gemm_dev(const bf* __restrict__ A, int lda,
                  const bf* __restrict__ Bm, int ldb,
                  void* __restrict__ C, int ldc,
                  const float* __restrict__ bias,
                  int K, float scale, int flags, int m0, int n0)
{
  __shared__ bf Btile[2][KSTEP * BLK_N];   // [k-run contiguous per n-row]
  const int tid  = threadIdx.x;
  const int lane = tid & 31;
  const int wave = tid >> 5;
  const int m_wave = m0 + wave * 32;

  f32x8 acc[2][4];
#pragma unroll
  for (int s = 0; s < 2; s++)
#pragma unroll
    for (int t2 = 0; t2 < 4; t2++)
#pragma unroll
      for (int r = 0; r < 8; r++) acc[s][t2][r] = 0.f;

  // staging: 256 threads x 16B = 4KB tile (64 n-rows x 32 k)
  const int sn = tid >> 2, sc = tid & 3;
  const size_t bfixed = (size_t)(n0 + sn) * ldb + sc * 8;
  const uint32_t ldse = (uint32_t)(sn * KSTEP + sc * 8) * 2;
  const uint32_t lds0 = (uint32_t)(uintptr_t)(&Btile[0][0]) + ldse;
  const uint32_t lds1 = (uint32_t)(uintptr_t)(&Btile[1][0]) + ldse;

  // prologue: stage first tile
  async_copy_b128((uint64_t)(uintptr_t)(Bm + bfixed), lds0);

  int cur = 0;
  for (int kk = 0; kk < K; kk += KSTEP) {
    wait_async();            // current tile landed in LDS
    __syncthreads();         // visible to all waves; prev buffer free
    if (kk + KSTEP < K)      // overlap next copy with this tile's math
      async_copy_b128((uint64_t)(uintptr_t)(Bm + bfixed + (kk + KSTEP)),
                      cur ? lds0 : lds1);

    const int kb = (lane >> 4) * 8;
    bf16x16 afrag[2];
#pragma unroll
    for (int s = 0; s < 2; s++) {
      const bf* ap = A + (size_t)(m_wave + s * 16 + (lane & 15)) * lda + kk + kb;
      uint4* u = reinterpret_cast<uint4*>(&afrag[s]);
      u[0] = *reinterpret_cast<const uint4*>(ap);
      u[1] = *reinterpret_cast<const uint4*>(ap + 16);
    }

    const bf* base = &Btile[cur][0];
#pragma unroll
    for (int t2 = 0; t2 < 4; t2++) {
      const bf* bp = base + (t2 * 16 + (lane & 15)) * KSTEP + kb;
      bf16x16 bfrag;
      uint4* u = reinterpret_cast<uint4*>(&bfrag);
      u[0] = *reinterpret_cast<const uint4*>(bp);
      u[1] = *reinterpret_cast<const uint4*>(bp + 16);
      acc[0][t2] = __builtin_amdgcn_wmma_f32_16x16x32_bf16(
          false, afrag[0], false, bfrag, (short)0, acc[0][t2], false, false);
      acc[1][t2] = __builtin_amdgcn_wmma_f32_16x16x32_bf16(
          false, afrag[1], false, bfrag, (short)0, acc[1][t2], false, false);
    }
    cur ^= 1;
  }

  // epilogue: C/D layout -> col = lane&15, vgpr r -> row = r + 8*(lane>>4)
#pragma unroll
  for (int s = 0; s < 2; s++) {
#pragma unroll
    for (int t2 = 0; t2 < 4; t2++) {
      const int cn = n0 + t2 * 16 + (lane & 15);
#pragma unroll
      for (int r = 0; r < 8; r++) {
        const int cm = m_wave + s * 16 + r + 8 * (lane >> 4);
        float v = acc[s][t2][r] * scale;
        if (flags & HAS_BIAS) v += bias[cn];
        if (flags & DO_GELU)  v = gelu_exact(v);
        const size_t idx = (flags & OUT_TRANS) ? (size_t)cn * ldc + cm
                                               : (size_t)cm * ldc + cn;
        if (flags & OUT_BF16) ((bf*)C)[idx] = (bf)v;
        else                  ((float*)C)[idx] = v;
      }
    }
  }
}

// ---------------- GEMM wrappers ----------------
__global__ __launch_bounds__(256)
void k_gemm(const bf* __restrict__ A, int lda, const bf* __restrict__ B, int ldb,
            void* __restrict__ C, int ldc, const float* __restrict__ bias,
            int K, float scale, int flags)
{
  gemm_dev(A, lda, B, ldb, C, ldc, bias, K, scale, flags,
           blockIdx.y * BLK_M, blockIdx.x * BLK_N);
}

// S[z] = Q_head @ K_head^T * (1/sqrt(HD)), z = b*NQ + h
__global__ __launch_bounds__(256)
void k_attn_scores(const bf* __restrict__ Q, const bf* __restrict__ Kbuf, bf* __restrict__ S)
{
  const int z = blockIdx.z, b = z / NQ_, h = z % NQ_, kv = h / (NQ_ / NKV_);
  const bf* A  = Q    + (size_t)b * NN * HID_ + (size_t)h * HD_;           // lda = HID
  const bf* Bm = Kbuf + (size_t)b * NN * (NKV_ * HD_) + (size_t)kv * HD_;  // (key,k), ldb=NKV*HD
  bf* C = S + (size_t)z * NN * NN;
  gemm_dev(A, HID_, Bm, NKV_ * HD_, C, NN, nullptr,
           HD_, 0.125f, OUT_BF16, blockIdx.y * BLK_M, blockIdx.x * BLK_N);
}

// O[:, h*HD:(h+1)*HD] = P[z] @ V_head, with V stored transposed (B, NKV*HD, NN)
__global__ __launch_bounds__(256)
void k_attn_pv(const bf* __restrict__ S, const bf* __restrict__ Vt, bf* __restrict__ O)
{
  const int z = blockIdx.z, b = z / NQ_, h = z % NQ_, kv = h / (NQ_ / NKV_);
  const bf* A  = S  + (size_t)z * NN * NN;                                  // lda = NN
  const bf* Bm = Vt + (size_t)b * (NKV_ * HD_) * NN + (size_t)kv * HD_ * NN; // (n,k): Vt[c][key]
  bf* C = O + (size_t)b * NN * HID_ + (size_t)h * HD_;                      // ldc = HID
  gemm_dev(A, NN, Bm, NN, C, HID_, nullptr,
           NN, 1.f, OUT_BF16, blockIdx.y * BLK_M, blockIdx.x * BLK_N);
}

// ---------------- softmax (in-place, row length NN) ----------------
__global__ __launch_bounds__(256)
void k_softmax(bf* __restrict__ S)
{
  __shared__ float red[256];
  bf* row = S + (size_t)blockIdx.x * NN;
  const int tid = threadIdx.x;
  float vals[4];
  float mx = -1e30f;
#pragma unroll
  for (int i = 0; i < 4; i++) { vals[i] = (float)row[tid + i * 256]; mx = fmaxf(mx, vals[i]); }
  red[tid] = mx; __syncthreads();
  for (int s = 128; s > 0; s >>= 1) { if (tid < s) red[tid] = fmaxf(red[tid], red[tid + s]); __syncthreads(); }
  mx = red[0]; __syncthreads();
  float sum = 0.f;
#pragma unroll
  for (int i = 0; i < 4; i++) { vals[i] = __expf(vals[i] - mx); sum += vals[i]; }
  red[tid] = sum; __syncthreads();
  for (int s = 128; s > 0; s >>= 1) { if (tid < s) red[tid] += red[tid + s]; __syncthreads(); }
  const float inv = 1.f / red[0];
#pragma unroll
  for (int i = 0; i < 4; i++) row[tid + i * 256] = (bf)(vals[i] * inv);
}

// ---------------- rmsnorm + adaLN modulate (one row per block) ----------------
__global__ __launch_bounds__(256)
void k_norm_mod(const float* __restrict__ x, const float* __restrict__ w,
                const float* __restrict__ mod, int sh_off, int sc_off,
                bf* __restrict__ out)
{
  __shared__ float red[256];
  const int row = blockIdx.x;          // row in [0, B*N)
  const int b   = row / NN;
  const int tid = threadIdx.x;
  const float* xr = x + (size_t)row * HID_;
  float v[4]; float ss = 0.f;
#pragma unroll
  for (int i = 0; i < 4; i++) { v[i] = xr[tid + i * 256]; ss += v[i] * v[i]; }
  red[tid] = ss; __syncthreads();
  for (int s = 128; s > 0; s >>= 1) { if (tid < s) red[tid] += red[tid + s]; __syncthreads(); }
  const float inv = rsqrtf(red[0] / (float)HID_ + EPS_);
#pragma unroll
  for (int i = 0; i < 4; i++) {
    const int c = tid + i * 256;
    float sc = 0.f, sh = 0.f;
    if (mod) { sc = mod[(size_t)b * 6 * HID_ + sc_off + c];
               sh = mod[(size_t)b * 6 * HID_ + sh_off + c]; }
    out[(size_t)row * HID_ + c] = (bf)(v[i] * inv * w[c] * (1.f + sc) + sh);
  }
}

// ---------------- gated residual add ----------------
__global__ __launch_bounds__(256)
void k_resid(float* __restrict__ x, const float* __restrict__ delta,
             const float* __restrict__ mod, int goff)
{
  const size_t idx = (size_t)blockIdx.x * 256 + threadIdx.x;    // < B*N*HID
  const int c = (int)(idx % HID_);
  const int b = (int)(idx / ((size_t)NN * HID_));
  x[idx] += mod[(size_t)b * 6 * HID_ + goff + c] * delta[idx];
}

// ---------------- RoPE, in-place on (B*N, nh, HD) bf16 ----------------
__global__ __launch_bounds__(256)
void k_rope(bf* __restrict__ X, int nh, int total)
{
  const size_t idx = (size_t)blockIdx.x * 256 + threadIdx.x;    // pair index
  if (idx >= (size_t)total) return;
  const int i = (int)(idx & 31);                 // 0..31
  const size_t r = idx >> 5;                     // (b*N + n)*nh + h
  const int nn2 = (int)((r / nh) % NN);
  const float invf = __expf(-(float)i * (logf(1e4f) / 32.f));   // 10000^(-2i/HD)
  float cs, sn; __sincosf((float)nn2 * invf, &sn, &cs);
  bf* p = X + r * HD_;
  const float x1 = (float)p[i], x2 = (float)p[i + 32];
  p[i]      = (bf)(x1 * cs - x2 * sn);
  p[i + 32] = (bf)(x2 * cs + x1 * sn);
}

// ---------------- patchify / unpatchify ----------------
__global__ __launch_bounds__(256)
void k_build_xin(const float* __restrict__ x, const float* __restrict__ cond, bf* __restrict__ xin)
{
  const size_t idx = (size_t)blockIdx.x * 256 + threadIdx.x;    // < B*N*8192
  const int q = (int)(idx % 8192);
  const size_t rn = idx / 8192;
  const int nn2 = (int)(rn % NN);
  const int b   = (int)(rn / NN);
  const int c = q >> 2, p = q & 3;
  float v;
  if (c < CIN_) v = x[((size_t)b * CIN_ + c) * T_ + nn2 * PP + p];
  else          v = cond[((size_t)b * CIN_ + (c - CIN_)) * T_ + nn2 * PP + p];
  xin[idx] = (bf)v;
}

__global__ __launch_bounds__(256)
void k_unpatch(const float* __restrict__ fin, float* __restrict__ out)
{
  const size_t idx = (size_t)blockIdx.x * 256 + threadIdx.x;    // < B*CIN*T
  const int tt = (int)(idx % T_);
  const size_t rc = idx / T_;
  const int c = (int)(rc % CIN_);
  const int b = (int)(rc / CIN_);
  const int nn2 = tt >> 2, p = tt & 3;
  out[idx] = fin[((size_t)b * NN + nn2) * (CIN_ * PP) + c * PP + p];
}

// ---------------- time embedding (tiny, one workgroup) ----------------
__global__ __launch_bounds__(256)
void k_time_embed(const float* __restrict__ t,
                  const float* __restrict__ w1, const float* __restrict__ b1,
                  const float* __restrict__ w2, const float* __restrict__ b2,
                  float* __restrict__ temb, float* __restrict__ stemb)
{
  __shared__ float e_[HID_];
  __shared__ float h_[HID_];
  const int tid = threadIdx.x;
  const float lsc = logf(1e4f) / 511.f;
  for (int b = 0; b < BB; b++) {
    const float tv = t[b];
    for (int j = tid; j < HID_; j += 256) {
      if (j < 512) e_[j] = sinf(tv * __expf(-(float)j * lsc));
      else         e_[j] = cosf(tv * __expf(-(float)(j - 512) * lsc));
    }
    __syncthreads();
    for (int j = tid; j < HID_; j += 256) {
      float acc = b1[j];
      for (int k = 0; k < HID_; k++) acc += e_[k] * w1[(size_t)k * HID_ + j];
      h_[j] = silu(acc);
    }
    __syncthreads();
    for (int j = tid; j < HID_; j += 256) {
      float acc = b2[j];
      for (int k = 0; k < HID_; k++) acc += h_[k] * w2[(size_t)k * HID_ + j];
      temb[(size_t)b * HID_ + j]  = acc;
      stemb[(size_t)b * HID_ + j] = silu(acc);
    }
    __syncthreads();
  }
}

// ---------------- adaLN: mod = silu(t_emb) @ ada_w + ada_b  (M=2, GEMV) ----------------
__global__ __launch_bounds__(256)
void k_ada(const float* __restrict__ st, const float* __restrict__ w,
           const float* __restrict__ bias, float* __restrict__ mod)
{
  const int j = blockIdx.x * 256 + threadIdx.x;   // < 6*HID
  for (int b = 0; b < BB; b++) {
    float acc = bias[j];
    for (int k = 0; k < HID_; k++) acc += st[b * HID_ + k] * w[(size_t)k * 6 * HID_ + j];
    mod[(size_t)b * 6 * HID_ + j] = acc;
  }
}

// ---------------- f32 -> bf16 with transpose: dst[n*K + k] = src[k*N + n] ----------------
__global__ __launch_bounds__(256)
void k_convert_t(const float* __restrict__ src, bf* __restrict__ dst, int K, int N)
{
  __shared__ bf tile[32][33];
  const int kb = blockIdx.x * 32, nb = blockIdx.y * 32;
  const int tx = threadIdx.x & 31, ty = threadIdx.x >> 5;   // 8 row-groups
  for (int i = ty; i < 32; i += 8)
    tile[i][tx] = (bf)src[(size_t)(kb + i) * N + nb + tx];
  __syncthreads();
  for (int i = ty; i < 32; i += 8)
    dst[(size_t)(nb + i) * K + kb + tx] = tile[tx][i];
}

// =============================================================================
extern "C" void kernel_launch(void* const* d_in, const int* in_sizes, int n_in,
                              void* d_out, int out_size, void* d_ws, size_t ws_size,
                              hipStream_t stream)
{
  (void)in_sizes; (void)n_in; (void)out_size; (void)ws_size;
  auto F = [&](int i) { return (const float*)d_in[i]; };
  const float* x     = F(0);
  const float* cond  = F(1);
  const float* t     = F(2);
  const float* pe_w1 = F(3);  const float* pe_b1 = F(4);
  const float* pe_w2 = F(5);  const float* pe_b2 = F(6);
  const float* te_w1 = F(7);  const float* te_b1 = F(8);
  const float* te_w2 = F(9);  const float* te_b2 = F(10);
  struct Blk { const float *n1,*n2,*wq,*wk,*wv,*wo,*w1,*b1,*w2,*b2,*aw,*ab; } blk[8];
  for (int d = 0; d < 8; d++) {
    const int s = 11 + d * 12;
    blk[d] = { F(s+0),F(s+1),F(s+2),F(s+3),F(s+4),F(s+5),F(s+6),F(s+7),F(s+8),F(s+9),F(s+10),F(s+11) };
  }
  const float* fin_n = F(107); const float* fin_w = F(108); const float* fin_b = F(109);

  // ---- workspace carve-up ----
  char* ws = (char*)d_ws; size_t off = 0;
  auto alloc = [&](size_t bytes) -> char* {
    char* p = ws + off; off += (bytes + 255) & ~(size_t)255; return p;
  };
  // transposed bf16 weights (N x K row-major)
  bf* tw_pe1 = (bf*)alloc((size_t)8192 * BOT_ * 2);
  bf* tw_pe2 = (bf*)alloc((size_t)BOT_ * HID_ * 2);
  bf* tw_fin = (bf*)alloc((size_t)HID_ * (CIN_ * PP) * 2);
  bf *tw_q[8], *tw_k[8], *tw_v[8], *tw_o[8], *tw_m1[8], *tw_m2[8];
  for (int d = 0; d < 8; d++) {
    tw_q[d]  = (bf*)alloc((size_t)HID_ * HID_ * 2);
    tw_k[d]  = (bf*)alloc((size_t)HID_ * (NKV_ * HD_) * 2);
    tw_v[d]  = (bf*)alloc((size_t)HID_ * (NKV_ * HD_) * 2);
    tw_o[d]  = (bf*)alloc((size_t)HID_ * HID_ * 2);
    tw_m1[d] = (bf*)alloc((size_t)HID_ * MLP_ * 2);
    tw_m2[d] = (bf*)alloc((size_t)MLP_ * HID_ * 2);
  }
  bf*    xin   = (bf*)alloc((size_t)BB * NN * 8192 * 2);
  bf*    h1    = (bf*)alloc((size_t)BB * NN * BOT_ * 2);
  float* xres  = (float*)alloc((size_t)BB * NN * HID_ * 4);
  bf*    xn    = (bf*)alloc((size_t)BB * NN * HID_ * 2);
  bf*    Qb    = (bf*)alloc((size_t)BB * NN * HID_ * 2);
  bf*    Kb    = (bf*)alloc((size_t)BB * NN * (NKV_ * HD_) * 2);
  bf*    Vt    = (bf*)alloc((size_t)BB * (NKV_ * HD_) * NN * 2);  // transposed V
  bf*    S     = (bf*)alloc((size_t)BB * NQ_ * NN * NN * 2);
  bf*    Ob    = (bf*)alloc((size_t)BB * NN * HID_ * 2);
  float* delta = (float*)alloc((size_t)BB * NN * HID_ * 4);
  bf*    mh    = (bf*)alloc((size_t)BB * NN * MLP_ * 2);
  float* finb  = (float*)alloc((size_t)BB * NN * (CIN_ * PP) * 4);
  float* temb  = (float*)alloc((size_t)BB * HID_ * 4);
  float* stemb = (float*)alloc((size_t)BB * HID_ * 4);
  float* mod   = (float*)alloc((size_t)BB * 6 * HID_ * 4);

  auto convT = [&](const float* s, bf* dptr, int K, int N) {
    k_convert_t<<<dim3(K / 32, N / 32), 256, 0, stream>>>(s, dptr, K, N);
  };
  auto gemm = [&](const bf* A, int lda, const bf* B, int ldb, void* C, int ldc,
                  const float* bias, int M, int Nn, int K, float scale, int flags) {
    dim3 g(Nn / BLK_N, M / BLK_M, 1);
    k_gemm<<<g, 256, 0, stream>>>(A, lda, B, ldb, C, ldc, bias, K, scale, flags);
  };

  // ---- weight conversion + transpose (bf16, N x K) ----
  convT(pe_w1, tw_pe1, 8192, BOT_);
  convT(pe_w2, tw_pe2, BOT_, HID_);
  convT(fin_w, tw_fin, HID_, CIN_ * PP);
  for (int d = 0; d < 8; d++) {
    convT(blk[d].wq, tw_q[d],  HID_, HID_);
    convT(blk[d].wk, tw_k[d],  HID_, NKV_ * HD_);
    convT(blk[d].wv, tw_v[d],  HID_, NKV_ * HD_);
    convT(blk[d].wo, tw_o[d],  HID_, HID_);
    convT(blk[d].w1, tw_m1[d], HID_, MLP_);
    convT(blk[d].w2, tw_m2[d], MLP_, HID_);
  }

  const int M = BB * NN;  // 2048 token rows

  // ---- patch embed ----
  k_build_xin<<<(int)(((size_t)M * 8192) / 256), 256, 0, stream>>>(x, cond, xin);
  gemm(xin, 8192, tw_pe1, 8192, h1, BOT_, pe_b1, M, BOT_, 8192, 1.f, OUT_BF16 | HAS_BIAS | DO_GELU);
  gemm(h1, BOT_, tw_pe2, BOT_, xres, HID_, pe_b2, M, HID_, BOT_, 1.f, HAS_BIAS);

  // ---- time embedding ----
  k_time_embed<<<1, 256, 0, stream>>>(t, te_w1, te_b1, te_w2, te_b2, temb, stemb);

  // ---- DiT blocks ----
  for (int d = 0; d < 8; d++) {
    k_ada<<<6 * HID_ / 256, 256, 0, stream>>>(stemb, blk[d].aw, blk[d].ab, mod);

    // attention branch
    k_norm_mod<<<M, 256, 0, stream>>>(xres, blk[d].n1, mod, 0, HID_, xn);
    gemm(xn, HID_, tw_q[d], HID_, Qb, HID_,       nullptr, M, HID_,       HID_, 1.f, OUT_BF16);
    gemm(xn, HID_, tw_k[d], HID_, Kb, NKV_ * HD_, nullptr, M, NKV_ * HD_, HID_, 1.f, OUT_BF16);
    for (int b = 0; b < BB; b++)   // V produced transposed, per batch
      gemm(xn + (size_t)b * NN * HID_, HID_, tw_v[d], HID_,
           Vt + (size_t)b * (NKV_ * HD_) * NN, NN, nullptr,
           NN, NKV_ * HD_, HID_, 1.f, OUT_BF16 | OUT_TRANS);
    k_rope<<<(M * NQ_ * 32) / 256, 256, 0, stream>>>(Qb, NQ_, M * NQ_ * 32);
    k_rope<<<(M * NKV_ * 32) / 256, 256, 0, stream>>>(Kb, NKV_, M * NKV_ * 32);
    k_attn_scores<<<dim3(NN / BLK_N, NN / BLK_M, BB * NQ_), 256, 0, stream>>>(Qb, Kb, S);
    k_softmax<<<BB * NQ_ * NN, 256, 0, stream>>>(S);
    k_attn_pv<<<dim3(1, NN / BLK_M, BB * NQ_), 256, 0, stream>>>(S, Vt, Ob);
    gemm(Ob, HID_, tw_o[d], HID_, delta, HID_, nullptr, M, HID_, HID_, 1.f, 0);
    k_resid<<<(M * HID_) / 256, 256, 0, stream>>>(xres, delta, mod, 2 * HID_);

    // MLP branch
    k_norm_mod<<<M, 256, 0, stream>>>(xres, blk[d].n2, mod, 3 * HID_, 4 * HID_, xn);
    gemm(xn, HID_, tw_m1[d], HID_, mh, MLP_, blk[d].b1, M, MLP_, HID_, 1.f, OUT_BF16 | HAS_BIAS | DO_GELU);
    gemm(mh, MLP_, tw_m2[d], MLP_, delta, HID_, blk[d].b2, M, HID_, MLP_, 1.f, HAS_BIAS);
    k_resid<<<(M * HID_) / 256, 256, 0, stream>>>(xres, delta, mod, 5 * HID_);
  }

  // ---- final projection + unpatchify ----
  k_norm_mod<<<M, 256, 0, stream>>>(xres, fin_n, nullptr, 0, 0, xn);
  gemm(xn, HID_, tw_fin, HID_, finb, CIN_ * PP, fin_b, M, CIN_ * PP, HID_, 1.f, HAS_BIAS);
  k_unpatch<<<(int)(((size_t)BB * CIN_ * T_) / 256), 256, 0, stream>>>(finb, (float*)d_out);
}
